// FlipTransformer_15058155339895
// MI455X (gfx1250) — matrix operand order repaired and checked by reading, not compiled, not run on hardware
//
#include <hip/hip_runtime.h>
#include <stdint.h>

// ---------------- model dims (match reference) ----------------
#define BN   2
#define SS   1024
#define DD   1024
#define HH   16
#define DHD  64
#define DFFN 4096
#define LLAY 2
#define VV   32000
#define GSZ  128
#define NT   (BN * SS)   // 2048 token rows

#define DI static __device__ __forceinline__

typedef __attribute__((ext_vector_type(16))) __bf16 v16bf;
typedef __attribute__((ext_vector_type(8)))  __bf16 v8bf;
typedef __attribute__((ext_vector_type(8)))  float  v8f;
typedef __attribute__((ext_vector_type(4))) unsigned int v4u;
typedef __attribute__((ext_vector_type(8)))  int v8i;
typedef __attribute__((ext_vector_type(4)))  int v4i;

DI __bf16 f2bf(float f) {
  uint32_t u = __builtin_bit_cast(uint32_t, f);
  uint32_t r = u + 0x7FFFu + ((u >> 16) & 1u);   // round-to-nearest-even
  uint16_t s = (uint16_t)(r >> 16);
  return __builtin_bit_cast(__bf16, s);
}
DI uint16_t f2bfu(float f) {
  uint32_t u = __builtin_bit_cast(uint32_t, f);
  uint32_t r = u + 0x7FFFu + ((u >> 16) & 1u);
  return (uint16_t)(r >> 16);
}
DI float bf2f(__bf16 h) {
  uint16_t s = __builtin_bit_cast(uint16_t, h);
  uint32_t u = ((uint32_t)s) << 16;
  return __builtin_bit_cast(float, u);
}

// ---------------- embedding: dequant ternary row + gather ----------------
__global__ void k_embed(const int* __restrict__ ids, const int8_t* __restrict__ et,
                        const float* __restrict__ es, float* __restrict__ x) {
  int n = blockIdx.x;
  int id = ids[n];
  size_t base = (size_t)id * DD;
  for (int d = threadIdx.x; d < DD; d += blockDim.x) {
    size_t idx = base + d;
    x[(size_t)n * DD + d] = (float)et[idx] * es[idx >> 7];  // GS=128
  }
}

// ---------------- RMSNorm -> bf16 ----------------
__global__ void k_rmsnorm(const float* __restrict__ x, const float* __restrict__ w,
                          __bf16* __restrict__ out) {
  __shared__ float red[256];
  int n = blockIdx.x;
  const float* xr = x + (size_t)n * DD;
  float ss = 0.f;
  for (int d = threadIdx.x; d < DD; d += 256) { float v = xr[d]; ss += v * v; }
  red[threadIdx.x] = ss;
  __syncthreads();
  for (int s = 128; s > 0; s >>= 1) {
    if (threadIdx.x < s) red[threadIdx.x] += red[threadIdx.x + s];
    __syncthreads();
  }
  float rs = rsqrtf(red[0] / (float)DD + 1e-6f);
  for (int d = threadIdx.x; d < DD; d += 256)
    out[(size_t)n * DD + d] = f2bf(xr[d] * rs * w[d]);
}

// ---------------- ternary-weight GEMM: C[M,O] = A[M,K](bf16) @ W[O,K].T ----------------
// block = 256 threads = 8 waves (2 x 4); tile 64(M) x 128(O); K chunk 32 via LDS.
// A tile staged by the Tensor Data Mover (tensor_load_to_lds, TENSORcnt);
// ternary W dequantized in-register (3-way select) and stored as one v16bf per thread.
enum { EPI_F32 = 0, EPI_BF16 = 1, EPI_ADDX = 2, EPI_SILU = 3 };

template <int EPI>
__global__ void k_gemm_tern(const __bf16* __restrict__ A, const int8_t* __restrict__ Wt,
                            const float* __restrict__ Wsc, int K, int O,
                            void* __restrict__ outp, const float* __restrict__ aux) {
  __shared__ __bf16 As[64 * 32];    // 4 KB, written by TDM
  __shared__ __bf16 Bs[128 * 32];   // 8 KB (dequantized W tile, row-major [o][k])
  const int tid  = threadIdx.x;
  const int lane = tid & 31, wid = tid >> 5;
  const int wr = wid >> 2, wc = wid & 3;
  const int m0 = blockIdx.x * 64, o0 = blockIdx.y * 128;

  v8f acc[2][2];
#pragma unroll
  for (int i = 0; i < 2; ++i)
#pragma unroll
    for (int j = 0; j < 2; ++j) acc[i][j] = (v8f){};

  const int brow = tid >> 1, bseg = tid & 1;   // W stage: 128 rows x 2 segs of 16 int8
  const int8_t* Wptr = Wt + (size_t)(o0 + brow) * K + bseg * 16;
  const size_t wbase = (size_t)(o0 + brow) * K + bseg * 16;

  // TDM descriptor pieces that are loop-invariant
  const uint32_t ldsA   = (uint32_t)(uintptr_t)(&As[0]);
  const uint64_t gbaseA = (uint64_t)(uintptr_t)(A + (size_t)m0 * K);

  const int klA = (lane < 16) ? 0 : 16;
  const int klB = (lane < 16) ? 0 : 16;
  const int nn  = lane & 15;

  for (int k0 = 0; k0 < K; k0 += 32) {
    __syncthreads();
    // ---- stage A tile 64x32 bf16 via Tensor Data Mover (wave 0 issues the DMA) ----
    if (wid == 0) {
      uint64_t ga = gbaseA + (uint64_t)k0 * 2u;   // byte address of tile start
      v4u g0;
      g0[0] = 1u;                                  // count=1, user mode
      g0[1] = ldsA;                                // lds_addr
      g0[2] = (uint32_t)ga;                        // global_addr[31:0]
      g0[3] = (uint32_t)(ga >> 32) | (2u << 30);   // global_addr[56:32] | type=2
      v8i g1;
      g1[0] = (int)(1u << 16);                     // data_size = 2 bytes
      g1[1] = (int)(((uint32_t)K & 0xFFFFu) << 16);          // tensor_dim0 lo16
      g1[2] = (int)(((uint32_t)K >> 16) | (64u << 16));      // dim0 hi16 | tensor_dim1 lo16 (=64)
      g1[3] = (int)(32u << 16);                    // tensor_dim1 hi16(0) | tile_dim0 = 32
      g1[4] = 64;                                  // tile_dim1 = 64, tile_dim2 = 0
      g1[5] = K;                                   // tensor_dim0_stride lo32 (= K elems)
      g1[6] = 0;                                   // stride0 hi16 | stride1 lo16
      g1[7] = 0;
      v4i g2 = (v4i){};
      v4i g3 = (v4i){};
      v8i g4 = (v8i){};
      __builtin_amdgcn_tensor_load_to_lds(g0, g1, g2, g3, g4, 0);
    }
    // ---- stage W: b128 int8 load, ternary 3-way-select dequant, one v16bf LDS store ----
    {
      int4 wv = *(const int4*)(Wptr + k0);
      float sc = Wsc[(wbase + (size_t)k0) >> 7];
      uint16_t pv = f2bfu(sc), nv = f2bfu(-sc);
      int vals[4] = {wv.x, wv.y, wv.z, wv.w};
      v16bf wb;
#pragma unroll
      for (int q = 0; q < 4; ++q)
#pragma unroll
        for (int j = 0; j < 4; ++j) {
          int c = (int)(int8_t)((vals[q] >> (8 * j)) & 0xFF);
          uint16_t r = (c == 0) ? (uint16_t)0 : ((c > 0) ? pv : nv);
          wb[q * 4 + j] = __builtin_bit_cast(__bf16, r);
        }
      *(v16bf*)&Bs[brow * 32 + bseg * 16] = wb;
    }
    if (k0 + 32 < K)                              // global_prefetch next weight chunk
      __builtin_prefetch(Wptr + k0 + 32, 0, 1);
    if (wid == 0) __builtin_amdgcn_s_wait_tensorcnt(0);
    __syncthreads();

    // ---- fragments from LDS ----
    v16bf afr[2], bfr[2];
#pragma unroll
    for (int mt = 0; mt < 2; ++mt) {
      int row = (wr * 32 + mt * 16 + nn) * 32;
      v8bf lo = *(const v8bf*)&As[row + klA];
      v8bf hi = *(const v8bf*)&As[row + 8 + klA];
#pragma unroll
      for (int j = 0; j < 8; ++j) { afr[mt][j] = lo[j]; afr[mt][8 + j] = hi[j]; }
    }
#pragma unroll
    for (int ot = 0; ot < 2; ++ot)
      bfr[ot] = *(const v16bf*)&Bs[(wc * 32 + ot * 16 + nn) * 32 + klB];

#pragma unroll
    for (int mt = 0; mt < 2; ++mt)
#pragma unroll
      for (int ot = 0; ot < 2; ++ot)
        acc[mt][ot] = __builtin_amdgcn_wmma_f32_16x16x32_bf16(
            false, afr[mt], false, bfr[ot], (short)0, acc[mt][ot], false, false);
  }

  // epilogue: C layout -> m = base + r + (lane<16?0:8), n = base + (lane&15)
  const int mh = (lane < 16) ? 0 : 8;
#pragma unroll
  for (int mt = 0; mt < 2; ++mt)
#pragma unroll
    for (int ot = 0; ot < 2; ++ot)
#pragma unroll
      for (int r = 0; r < 8; ++r) {
        int m = m0 + wr * 32 + mt * 16 + r + mh;
        int n = o0 + wc * 32 + ot * 16 + nn;
        size_t idx = (size_t)m * O + n;
        float v = acc[mt][ot][r];
        if (EPI == EPI_F32) {
          ((float*)outp)[idx] = v;
        } else if (EPI == EPI_BF16) {
          ((__bf16*)outp)[idx] = f2bf(v);
        } else if (EPI == EPI_ADDX) {
          ((float*)outp)[idx] = aux[idx] + v;      // residual add into x
        } else {                                    // EPI_SILU: silu(gate)*up -> bf16
          float g = aux[idx];
          float sg = g / (1.f + __expf(-g));
          ((__bf16*)outp)[idx] = f2bf(sg * v);
        }
      }
}

// ---------------- V transpose: [B,S,H,DH] -> [B,H,DH,S] (bf16) ----------------
__global__ void k_vtrans(const __bf16* __restrict__ v, __bf16* __restrict__ vT) {
  int n = blockIdx.x;
  int b = n / SS, t = n % SS;
  for (int idx = threadIdx.x; idx < DD; idx += 256) {
    int h = idx >> 6, d = idx & 63;
    vT[((size_t)(b * HH + h) * DHD + d) * SS + t] = v[(size_t)n * DD + idx];
  }
}

// ---------------- flash attention: one wave per (b, h, 16-row query tile) ----------------
__global__ void k_attn(const __bf16* __restrict__ qb, const __bf16* __restrict__ kb,
                       const __bf16* __restrict__ vT, const __bf16* __restrict__ hn,
                       const float* __restrict__ alpha, __bf16* __restrict__ ao) {
  __shared__ __bf16 Plds[8][16 * 32];   // per-wave P staging (C-layout -> A-layout)
  const int lane = threadIdx.x & 31, wid = threadIdx.x >> 5;
  const int gidx = blockIdx.x * 8 + wid;
  const int QT = SS / 16;
  const int s0 = (gidx % QT) * 16;
  const int bh = gidx / QT;
  const int h = bh % HH, b = bh / HH;

  const int nn  = lane & 15;
  const int mh  = (lane < 16) ? 0 : 8;
  const int klA = (lane < 16) ? 0 : 16;
  const int klB = (lane < 16) ? 0 : 16;

  // Q fragments (16 x 64 = two 16x32 A fragments)
  v16bf qf[2];
  {
    const __bf16* qrow = qb + (size_t)(b * SS + s0 + nn) * DD + h * DHD;
#pragma unroll
    for (int kk = 0; kk < 2; ++kk) {
      v8bf lo = *(const v8bf*)(qrow + kk * 32 + klA);
      v8bf hi = *(const v8bf*)(qrow + kk * 32 + 8 + klA);
#pragma unroll
      for (int j = 0; j < 8; ++j) { qf[kk][j] = lo[j]; qf[kk][8 + j] = hi[j]; }
    }
  }

  v8f oacc[4];
#pragma unroll
  for (int i = 0; i < 4; ++i) oacc[i] = (v8f){};
  float mrow[8], srow[8];
#pragma unroll
  for (int r = 0; r < 8; ++r) { mrow[r] = -1e30f; srow[r] = 0.f; }
  const float scale = 0.125f;   // DH^-0.5

  for (int t0 = 0; t0 <= s0 + 15; t0 += 32) {
    // ---- scores: two 16x16 tiles (keys t0..t0+31) ----
    v8f sc[2];
#pragma unroll
    for (int j = 0; j < 2; ++j) {
      v8f a = (v8f){};
      int tcol = t0 + j * 16 + nn;
      const __bf16* krow = kb + (size_t)(b * SS + tcol) * DD + h * DHD;
#pragma unroll
      for (int kk = 0; kk < 2; ++kk) {
        v16bf bf = *(const v16bf*)(krow + kk * 32 + klB);
        a = __builtin_amdgcn_wmma_f32_16x16x32_bf16(false, qf[kk], false, bf,
                                                    (short)0, a, false, false);
      }
      sc[j] = a;
    }
    // ---- causal mask + online softmax (row = r+mh across the 16 lanes of a half) ----
#pragma unroll
    for (int r = 0; r < 8; ++r) {
      int m = s0 + r + mh;
      float v0 = sc[0][r] * scale; if (t0 + nn > m)      v0 = -1e30f;
      float v1 = sc[1][r] * scale; if (t0 + 16 + nn > m) v1 = -1e30f;
      float cm = fmaxf(v0, v1);
      cm = fmaxf(cm, __shfl_xor(cm, 1, 32));
      cm = fmaxf(cm, __shfl_xor(cm, 2, 32));
      cm = fmaxf(cm, __shfl_xor(cm, 4, 32));
      cm = fmaxf(cm, __shfl_xor(cm, 8, 32));
      float mnew = fmaxf(mrow[r], cm);
      float corr = __expf(mrow[r] - mnew);
      float p0 = __expf(v0 - mnew), p1 = __expf(v1 - mnew);
      float rs = p0 + p1;
      rs += __shfl_xor(rs, 1, 32);
      rs += __shfl_xor(rs, 2, 32);
      rs += __shfl_xor(rs, 4, 32);
      rs += __shfl_xor(rs, 8, 32);
      srow[r] = srow[r] * corr + rs;
      mrow[r] = mnew;
#pragma unroll
      for (int d = 0; d < 4; ++d) oacc[d][r] *= corr;
      Plds[wid][(r + mh) * 32 + nn]      = f2bf(p0);
      Plds[wid][(r + mh) * 32 + 16 + nn] = f2bf(p1);
    }
    // ---- P fragment (A-layout) from LDS ----
    v16bf pf;
    {
      const __bf16* pr = &Plds[wid][nn * 32];
      v8bf lo = *(const v8bf*)(pr + klA);
      v8bf hi = *(const v8bf*)(pr + 8 + klA);
#pragma unroll
      for (int j = 0; j < 8; ++j) { pf[j] = lo[j]; pf[8 + j] = hi[j]; }
    }
    // ---- P @ V : V^T gives contiguous B fragments along t ----
#pragma unroll
    for (int ds = 0; ds < 4; ++ds) {
      const __bf16* vr = vT + ((size_t)(b * HH + h) * DHD + ds * 16 + nn) * SS + t0 + klB;
      v16bf vf = *(const v16bf*)vr;
      oacc[ds] = __builtin_amdgcn_wmma_f32_16x16x32_bf16(false, pf, false, vf,
                                                         (short)0, oacc[ds], false, false);
    }
  }

  // ---- normalize, add alpha * h_heads, store bf16 ----
  float al = alpha[h];
#pragma unroll
  for (int ds = 0; ds < 4; ++ds)
#pragma unroll
    for (int r = 0; r < 8; ++r) {
      int m = s0 + r + mh;
      int d = ds * 16 + nn;
      size_t idx = (size_t)(b * SS + m) * DD + h * DHD + d;
      float v = oacc[ds][r] / srow[r] + al * bf2f(hn[idx]);
      ao[idx] = f2bf(v);
    }
}

// ---------------- host-side orchestration ----------------
extern "C" void kernel_launch(void* const* d_in, const int* in_sizes, int n_in,
                              void* d_out, int out_size, void* d_ws, size_t ws_size,
                              hipStream_t stream) {
  (void)in_sizes; (void)n_in; (void)out_size; (void)ws_size;

  const int*    ids   = (const int*)d_in[0];
  const int8_t* emb_t = (const int8_t*)d_in[1];
  const float*  emb_s = (const float*)d_in[2];
  const int8_t* qt = (const int8_t*)d_in[3];   const float* qs  = (const float*)d_in[4];
  const int8_t* kt = (const int8_t*)d_in[5];   const float* ksc = (const float*)d_in[6];
  const int8_t* vt = (const int8_t*)d_in[7];   const float* vsc = (const float*)d_in[8];
  const int8_t* owt = (const int8_t*)d_in[9];  const float* osc = (const float*)d_in[10];
  const int8_t* gwt = (const int8_t*)d_in[11]; const float* gsc = (const float*)d_in[12];
  const int8_t* uwt = (const int8_t*)d_in[13]; const float* usc = (const float*)d_in[14];
  const int8_t* dwt = (const int8_t*)d_in[15]; const float* dsc = (const float*)d_in[16];
  const float* wa    = (const float*)d_in[17];
  const float* wm    = (const float*)d_in[18];
  const float* alpha = (const float*)d_in[19];
  const float* wf    = (const float*)d_in[20];
  const int8_t* hwt  = (const int8_t*)d_in[21];
  const float*  hsc  = (const float*)d_in[22];
  float* logits = (float*)d_out;

  // workspace carve-out (~80 MB total)
  char* wsp = (char*)d_ws;
  size_t off = 0;
  auto alloc = [&](size_t bytes) -> void* {
    void* p = wsp + off;
    off = (off + bytes + 255) & ~(size_t)255;
    return p;
  };
  float*  x   = (float*)alloc((size_t)NT * DD * 4);
  __bf16* hn  = (__bf16*)alloc((size_t)NT * DD * 2);
  __bf16* qb  = (__bf16*)alloc((size_t)NT * DD * 2);
  __bf16* kb  = (__bf16*)alloc((size_t)NT * DD * 2);
  __bf16* vb  = (__bf16*)alloc((size_t)NT * DD * 2);
  __bf16* vTb = (__bf16*)alloc((size_t)BN * HH * DHD * SS * 2);
  __bf16* aob = (__bf16*)alloc((size_t)NT * DD * 2);
  float*  g   = (float*)alloc((size_t)NT * DFFN * 4);
  __bf16* gu  = (__bf16*)alloc((size_t)NT * DFFN * 2);

  const int ngdd = DD * DD / GSZ;      // 8192  scales per D x D weight
  const int ngfd = DFFN * DD / GSZ;    // 32768 scales per DFF x D weight

  dim3 blk(256);
  dim3 gD(NT / 64, DD / 128);          // GEMM grid for O = D
  dim3 gF(NT / 64, DFFN / 128);        // GEMM grid for O = DFF
  dim3 gV(NT / 64, VV / 128);          // GEMM grid for O = V

  k_embed<<<NT, blk, 0, stream>>>(ids, emb_t, emb_s, x);

  for (int l = 0; l < LLAY; ++l) {
    const size_t wdd = (size_t)l * DD * DD;
    const size_t wfd = (size_t)l * DFFN * DD;

    // attention block
    k_rmsnorm<<<NT, blk, 0, stream>>>(x, wa + l * DD, hn);
    k_gemm_tern<EPI_BF16><<<gD, blk, 0, stream>>>(hn, qt + wdd, qs + (size_t)l * ngdd, DD, DD, qb, nullptr);
    k_gemm_tern<EPI_BF16><<<gD, blk, 0, stream>>>(hn, kt + wdd, ksc + (size_t)l * ngdd, DD, DD, kb, nullptr);
    k_gemm_tern<EPI_BF16><<<gD, blk, 0, stream>>>(hn, vt + wdd, vsc + (size_t)l * ngdd, DD, DD, vb, nullptr);
    k_vtrans<<<NT, blk, 0, stream>>>(vb, vTb);
    k_attn<<<(BN * HH * (SS / 16)) / 8, blk, 0, stream>>>(qb, kb, vTb, hn, alpha + l * HH, aob);
    k_gemm_tern<EPI_ADDX><<<gD, blk, 0, stream>>>(aob, owt + wdd, osc + (size_t)l * ngdd, DD, DD, x, x);

    // SwiGLU MLP block
    k_rmsnorm<<<NT, blk, 0, stream>>>(x, wm + l * DD, hn);
    k_gemm_tern<EPI_F32><<<gF, blk, 0, stream>>>(hn, gwt + wfd, gsc + (size_t)l * ngfd, DD, DFFN, g, nullptr);
    k_gemm_tern<EPI_SILU><<<gF, blk, 0, stream>>>(hn, uwt + wfd, usc + (size_t)l * ngfd, DD, DFFN, gu, g);
    k_gemm_tern<EPI_ADDX><<<gD, blk, 0, stream>>>(gu, dwt + wfd, dsc + (size_t)l * ngfd, DFFN, DD, x, x);
  }

  // final norm + logits head
  k_rmsnorm<<<NT, blk, 0, stream>>>(x, wf, hn);
  k_gemm_tern<EPI_F32><<<gV, blk, 0, stream>>>(hn, hwt, hsc, DD, VV, logits, nullptr);
}
